// FeatureAgg_27401891348480
// MI455X (gfx1250) — compile-verified, matrix operand order checked by reading, not run on hardware
//
#include <hip/hip_runtime.h>

// Sizes from the reference
constexpr int NB = 4096;   // batch
constexpr int NK = 64;     // neighbors per node
constexpr int NE = 128;    // embedding dim
constexpr int NT = 3;      // types
constexpr int NN = 100000; // node table size

typedef __attribute__((ext_vector_type(16))) _Float16 v16h;
typedef __attribute__((ext_vector_type(8)))  float    v8f;
typedef __attribute__((ext_vector_type(4)))  float    f32x4;  // native vec4 (no ctors)

// Explicit AMDGPU address spaces: 1 = global, 3 = LDS.
typedef __attribute__((address_space(3))) _Float16 lh16;   // LDS half
typedef __attribute__((address_space(1))) _Float16 gh16;   // global half
typedef __attribute__((address_space(1))) float    gf32;   // global float
typedef __attribute__((address_space(1))) const f32x4 gcf4;
typedef __attribute__((address_space(1))) const int   gci32;

__device__ inline const lh16* to_lds(const _Float16* p) { return (const lh16*)p; }
__device__ inline       lh16* to_lds(      _Float16* p) { return (lh16*)p; }

// ---------------------------------------------------------------------------
// 64x128 GEMM tile engine: C[64 x 128] = relu(A[64 x K] @ B[K x 128] + bias)
// A: f16 row-major stride SA in LDS. B passed TRANSPOSED: BT[128 x K] f16 in
// global (L2-resident). 8 waves: wave w -> m-strip (w&3), n-tiles (w>>2)*4..+3.
// Fragment layout per CDNA5 ISA 16-bit operand tables:
//   lane<16 : rows r=lane,    K-chunks {k0..k0+7} (v0-3), {k0+16..k0+23} (v4-7)
//   lane>=16: rows r=lane-16, K-chunks {k0+8..k0+15},     {k0+24..k0+31}
// ---------------------------------------------------------------------------
template<int KSTEPS, bool RELU, typename OutPtr>
__device__ inline void wmma_gemm64(const lh16* A, int SA,
                                   const gh16* BT,
                                   const gf32* bias,
                                   OutPtr Out, int SO)
{
  constexpr int KK = KSTEPS * 32;
  const int tid  = threadIdx.x;
  const int wave = tid >> 5;
  const int lane = tid & 31;
  const int mt   = wave & 3;          // m tile 0..3
  const int nt0  = (wave >> 2) << 2;  // first n tile (0 or 4)
  const int sub  = (lane >> 4) << 3;  // 0 or 8: k-chunk offset / m offset
  const int rlo  = lane & 15;

  // preload all A fragments for this wave's m-strip
  v16h afrag[KSTEPS];
  {
    const lh16* ap = A + ((mt << 4) + rlo) * SA + sub;
#pragma unroll
    for (int ks = 0; ks < KSTEPS; ks++) {
      const lh16* p = ap + (ks << 5);
      v16h f;
#pragma unroll
      for (int i = 0; i < 8; i++) { f[i] = p[i]; f[8 + i] = p[16 + i]; }
      afrag[ks] = f;
    }
  }

#pragma unroll
  for (int nt = nt0; nt < nt0 + 4; nt++) {
    v8f acc = {};
    const gh16* bp0 = BT + (size_t)((nt << 4) + rlo) * KK + sub;
#pragma unroll
    for (int ks = 0; ks < KSTEPS; ks++) {
      const gh16* p = bp0 + (ks << 5);
      v16h bf;
#pragma unroll
      for (int i = 0; i < 8; i++) { bf[i] = p[i]; bf[8 + i] = p[16 + i]; }
      acc = __builtin_amdgcn_wmma_f32_16x16x32_f16(
          false, afrag[ks], false, bf, (short)0, acc, false, false);
    }
    const int n    = (nt << 4) + rlo;
    const float bv = bias[n];
    const int mbase = (mt << 4) + sub;
#pragma unroll
    for (int r = 0; r < 8; r++) {
      float v = acc[r] + bv;
      if (RELU) v = fmaxf(v, 0.0f);
      Out[(size_t)(mbase + r) * SO + n] = v;   // converts to pointee type
    }
  }
}

// ---------------------------------------------------------------------------
// Kernel 0: transpose + f32->f16 convert all GEMM weights into workspace.
// WfT/Wa1T/WcT: [128][256], Wa2T/W2T: [128][128]
// ---------------------------------------------------------------------------
__global__ void prep_kernel(const float* W_f_, const float* Wa1_,
                            const float* Wa2_, const float* W2_,
                            const float* Wc_,
                            _Float16* WfT_, _Float16* Wa1T_,
                            _Float16* Wa2T_, _Float16* W2T_,
                            _Float16* WcT_)
{
  const gf32* W_f = (const gf32*)W_f_;  const gf32* Wa1 = (const gf32*)Wa1_;
  const gf32* Wa2 = (const gf32*)Wa2_;  const gf32* W2  = (const gf32*)W2_;
  const gf32* Wc  = (const gf32*)Wc_;
  gh16* WfT  = (gh16*)WfT_;  gh16* Wa1T = (gh16*)Wa1T_;
  gh16* Wa2T = (gh16*)Wa2T_; gh16* W2T  = (gh16*)W2T_;
  gh16* WcT  = (gh16*)WcT_;

  int i = blockIdx.x * 256 + threadIdx.x;          // 0 .. 32767
  if (i < 128 * 256) {
    int n = i >> 8, k = i & 255;                   // out[n][k] = in[k][n]
    WfT[i]  = (_Float16)W_f[k * 128 + n];
    Wa1T[i] = (_Float16)Wa1[k * 128 + n];
    WcT[i]  = (_Float16)Wc [k * 128 + n];
  }
  if (i < 128 * 128) {
    int n = i >> 7, k = i & 127;
    Wa2T[i] = (_Float16)Wa2[k * 128 + n];
    W2T[i]  = (_Float16)W2 [k * 128 + n];
  }
}

// ---------------------------------------------------------------------------
// Kernel 1: nodes_fusion = relu([node_emb[nodes] | node_prof[nodes]] @ W_f + b_f)
// One block = 64 batch rows. Output stored f16 (consumed only by later GEMMs).
// ---------------------------------------------------------------------------
__global__ __launch_bounds__(256)
void fusion_kernel(const int* nodes_,
                   const float* node_emb, const float* node_prof,
                   const _Float16* WfT, const float* b_f,
                   _Float16* nf16)
{
  const int b0 = blockIdx.x << 6;
  const int tid = threadIdx.x;
  __shared__ __align__(16) _Float16 bufX[64 * 256];
  __shared__ int sIdx[64];

  if (tid < 64) sIdx[tid] = ((gci32*)nodes_)[b0 + tid];
  __syncthreads();

  for (int i = tid; i < 64 * 64; i += 256) {       // 16B-granular gather
    int row = i >> 6, q = i & 63;
    int node = sIdx[row];
    f32x4 v = (q < 32)
        ? ((gcf4*)(node_emb  + (size_t)node * NE))[q]
        : ((gcf4*)(node_prof + (size_t)node * NE))[q - 32];
    lh16* dst = to_lds(&bufX[row * 256 + q * 4]);
    dst[0] = (_Float16)v[0]; dst[1] = (_Float16)v[1];
    dst[2] = (_Float16)v[2]; dst[3] = (_Float16)v[3];
  }
  __syncthreads();
  wmma_gemm64<8, true>(to_lds(bufX), 256, (const gh16*)WfT, (const gf32*)b_f,
                       (gh16*)(nf16 + (size_t)b0 * NE), NE);
}

// ---------------------------------------------------------------------------
// Kernel 2: per (b, t): the full attention pipeline over 64 neighbors.
// ---------------------------------------------------------------------------
__global__ __launch_bounds__(256)
void type_kernel(const int* neigh_idx,
                 const float* neigh_emb, const float* neigh_prof,
                 const _Float16* WfT, const float* b_f,
                 const _Float16* Wa1T, const float* ba1,
                 const _Float16* Wa2T, const float* ba2,
                 const float* Wa3_, const float* ba3_,
                 const float* W1_, const float* b1_,
                 const _Float16* nf16_,
                 float* typeagg_)
{
  const int b = blockIdx.x;          // 0..B-1
  const int t = blockIdx.y;          // 0..T-1
  const int tid = threadIdx.x;

  __shared__ __align__(16) _Float16 bufX [64 * 256]; // gathered [e|p]; reused for h2
  __shared__ __align__(16) _Float16 bufX2[64 * 256]; // [nf | nodes_fusion]
  __shared__ __align__(16) _Float16 bufH [64 * 128]; // h1
  __shared__ int   sIdx[64];
  __shared__ float sLogit[64];
  __shared__ float sAtt[64];
  __shared__ float sAgg[128];

  const gf32* Wa3 = (const gf32*)Wa3_;
  const gf32* ba3 = (const gf32*)ba3_;
  const gf32* W1  = (const gf32*)W1_;
  const gf32* b1  = (const gf32*)b1_;
  const gh16* nf16 = (const gh16*)nf16_;
  gf32* typeagg = (gf32*)typeagg_;

  if (tid < 64) sIdx[tid] = ((gci32*)neigh_idx)[((size_t)t * NB + b) * NK + tid];
  __syncthreads();

  const float* embT  = neigh_emb  + (size_t)t * NN * NE;
  const float* profT = neigh_prof + (size_t)t * NN * NE;
  for (int i = tid; i < 64 * 64; i += 256) {
    int row = i >> 6, q = i & 63;
    int node = sIdx[row];
    f32x4 v = (q < 32)
        ? ((gcf4*)(embT  + (size_t)node * NE))[q]
        : ((gcf4*)(profT + (size_t)node * NE))[q - 32];
    lh16* dst = to_lds(&bufX[row * 256 + q * 4]);
    dst[0] = (_Float16)v[0]; dst[1] = (_Float16)v[1];
    dst[2] = (_Float16)v[2]; dst[3] = (_Float16)v[3];
  }
  // broadcast nodes_fusion[b] into bufX2 cols 128..255
  for (int i = tid; i < 64 * 128; i += 256) {
    int row = i >> 7, col = i & 127;
    to_lds(bufX2)[row * 256 + 128 + col] = nf16[(size_t)b * NE + col];
  }
  __syncthreads();

  // nf = relu([e|p] @ W_f + b_f) -> bufX2 cols 0..127
  wmma_gemm64<8, true>(to_lds(bufX), 256, (const gh16*)WfT, (const gf32*)b_f,
                       to_lds(bufX2), 256);
  __syncthreads();
  // h1 = relu([nf|fusion] @ Wa1 + ba1)
  wmma_gemm64<8, true>(to_lds(bufX2), 256, (const gh16*)Wa1T, (const gf32*)ba1,
                       to_lds(bufH), 128);
  __syncthreads();
  // h2 = relu(h1 @ Wa2 + ba2) -> reuse bufX (first 64x128)
  wmma_gemm64<4, true>(to_lds(bufH), 128, (const gh16*)Wa2T, (const gf32*)ba2,
                       to_lds(bufX), 128);
  __syncthreads();

  // attention logits: h2 @ Wa3 + ba3  (one row per thread)
  if (tid < 64) {
    float acc = ba3[0];
    for (int c = 0; c < 128; c++) acc += (float)bufX[tid * 128 + c] * Wa3[c];
    sLogit[tid] = acc;
  }
  __syncthreads();
  if (tid < 64) {
    float m = -1e30f;
    for (int j = 0; j < 64; j++) m = fmaxf(m, sLogit[j]);
    float s = 0.f;
    for (int j = 0; j < 64; j++) s += __expf(sLogit[j] - m);
    sAtt[tid] = __expf(sLogit[tid] - m) / s;
  }
  __syncthreads();

  // agg[c] = sum_k att[k] * nf[k][c]
  if (tid < 128) {
    float acc = 0.f;
    for (int k = 0; k < 64; k++) acc += sAtt[k] * (float)bufX2[k * 256 + tid];
    sAgg[tid] = acc;
  }
  __syncthreads();

  // per_type = relu(agg @ W1 + b1) -> typeagg[b][t*128 + c]
  if (tid < 128) {
    float acc = b1[tid];
    for (int d = 0; d < 128; d++) acc += sAgg[d] * W1[d * 128 + tid];
    typeagg[(size_t)b * (NT * NE) + t * NE + tid] = fmaxf(acc, 0.f);
  }
}

// ---------------------------------------------------------------------------
// Kernel 3: type softmax, weighted type sum, W2, concat, Wc, outputs.
// One block = 64 batch rows.
// ---------------------------------------------------------------------------
__global__ __launch_bounds__(256)
void final_kernel(const float* typeagg_, const float* Wt_,
                  const _Float16* W2T, const float* b2,
                  const _Float16* WcT, const float* bc,
                  const _Float16* nf16_,
                  float* out_, float* att_out_)
{
  const int b0 = blockIdx.x << 6;
  const int tid = threadIdx.x;
  __shared__ float sAttT[64 * 3];
  __shared__ __align__(16) _Float16 bufF[64 * 128];
  __shared__ __align__(16) _Float16 bufC[64 * 256];

  const gf32* typeagg = (const gf32*)typeagg_;
  const gf32* Wt = (const gf32*)Wt_;
  const gh16* nf16 = (const gh16*)nf16_;
  gf32* att_out = (gf32*)att_out_;

  if (tid < 64) {
    const gf32* ta = typeagg + (size_t)(b0 + tid) * (NT * NE);
    float l0 = 0.f, l1 = 0.f, l2 = 0.f;
    for (int j = 0; j < NT * NE; j++) {
      float v = ta[j];
      l0 += v * Wt[j * 3 + 0]; l1 += v * Wt[j * 3 + 1]; l2 += v * Wt[j * 3 + 2];
    }
    float m = fmaxf(l0, fmaxf(l1, l2));
    float e0 = __expf(l0 - m), e1 = __expf(l1 - m), e2 = __expf(l2 - m);
    float s = e0 + e1 + e2;
    float a0 = e0 / s, a1 = e1 / s, a2 = e2 / s;
    sAttT[tid * 3 + 0] = a0; sAttT[tid * 3 + 1] = a1; sAttT[tid * 3 + 2] = a2;
    att_out[(size_t)(b0 + tid) * 3 + 0] = a0;
    att_out[(size_t)(b0 + tid) * 3 + 1] = a1;
    att_out[(size_t)(b0 + tid) * 3 + 2] = a2;
  }
  __syncthreads();

  for (int i = tid; i < 64 * 128; i += 256) {
    int row = i >> 7, col = i & 127;
    const gf32* ta = typeagg + (size_t)(b0 + row) * (NT * NE);
    float v = sAttT[row * 3 + 0] * ta[col] +
              sAttT[row * 3 + 1] * ta[128 + col] +
              sAttT[row * 3 + 2] * ta[256 + col];
    to_lds(bufF)[row * 128 + col] = (_Float16)v;
    to_lds(bufC)[row * 256 + col] = nf16[(size_t)(b0 + row) * NE + col]; // concat lhs
  }
  __syncthreads();
  // final = relu(final0 @ W2 + b2) -> bufC cols 128..255
  wmma_gemm64<4, true>(to_lds(bufF), 128, (const gh16*)W2T, (const gf32*)b2,
                       to_lds(bufC) + 128, 256);
  __syncthreads();
  // out = relu([fusion|final] @ Wc + bc)
  wmma_gemm64<8, true>(to_lds(bufC), 256, (const gh16*)WcT, (const gf32*)bc,
                       (gf32*)(out_ + (size_t)b0 * NE), NE);
}

// ---------------------------------------------------------------------------
extern "C" void kernel_launch(void* const* d_in, const int* in_sizes, int n_in,
                              void* d_out, int out_size, void* d_ws, size_t ws_size,
                              hipStream_t stream)
{
  (void)in_sizes; (void)n_in; (void)out_size; (void)ws_size;
  const int*   nodes      = (const int*)  d_in[0];
  const int*   neigh_idx  = (const int*)  d_in[1];
  const float* node_emb   = (const float*)d_in[2];
  const float* node_prof  = (const float*)d_in[3];
  const float* neigh_emb  = (const float*)d_in[4];
  const float* neigh_prof = (const float*)d_in[5];
  const float* W_f = (const float*)d_in[6];   const float* b_f = (const float*)d_in[7];
  const float* Wa1 = (const float*)d_in[8];   const float* ba1 = (const float*)d_in[9];
  const float* Wa2 = (const float*)d_in[10];  const float* ba2 = (const float*)d_in[11];
  const float* Wa3 = (const float*)d_in[12];  const float* ba3 = (const float*)d_in[13];
  const float* W1  = (const float*)d_in[14];  const float* b1  = (const float*)d_in[15];
  const float* W2  = (const float*)d_in[16];  const float* b2  = (const float*)d_in[17];
  const float* Wc  = (const float*)d_in[18];  const float* bc  = (const float*)d_in[19];
  const float* Wt  = (const float*)d_in[20];

  char* ws = (char*)d_ws;
  _Float16* WfT  = (_Float16*)ws;  ws += 128 * 256 * 2;
  _Float16* Wa1T = (_Float16*)ws;  ws += 128 * 256 * 2;
  _Float16* Wa2T = (_Float16*)ws;  ws += 128 * 128 * 2;
  _Float16* W2T  = (_Float16*)ws;  ws += 128 * 128 * 2;
  _Float16* WcT  = (_Float16*)ws;  ws += 128 * 256 * 2;
  _Float16* nf16 = (_Float16*)ws;  ws += (size_t)NB * NE * 2;
  float* typeagg = (float*)ws;     ws += (size_t)NB * NT * NE * 4;

  float* out     = (float*)d_out;
  float* att_out = out + (size_t)NB * NE;

  prep_kernel<<<dim3(128), dim3(256), 0, stream>>>(
      W_f, Wa1, Wa2, W2, Wc, WfT, Wa1T, Wa2T, W2T, WcT);
  fusion_kernel<<<dim3(NB / 64), dim3(256), 0, stream>>>(
      nodes, node_emb, node_prof, WfT, b_f, nf16);
  type_kernel<<<dim3(NB, NT), dim3(256), 0, stream>>>(
      neigh_idx, neigh_emb, neigh_prof, WfT, b_f, Wa1T, ba1, Wa2T, ba2,
      Wa3, ba3, W1, b1, nf16, typeagg);
  final_kernel<<<dim3(NB / 64), dim3(256), 0, stream>>>(
      typeagg, Wt, W2T, b2, WcT, bc, nf16, out, att_out);
}
